// SAGE_52905407152435
// MI455X (gfx1250) — matrix operand order repaired and checked by reading, not compiled, not run on hardware
//
#include <hip/hip_runtime.h>
#include <hip/hip_bf16.h>
#include <math.h>

// ---------- problem constants (match reference) ----------
#define NN     150000
#define T1     30000
#define T2     8192
#define E1     400000
#define E2     160000
#define F_IN   512
#define F_HID  256
#define F_OUT  20
#define F_OUT_PAD 32

typedef __bf16 bf16_t;
typedef bf16_t v16bf __attribute__((ext_vector_type(16)));
typedef bf16_t v8bf  __attribute__((ext_vector_type(8)));
typedef float  v8f   __attribute__((ext_vector_type(8)));

// ------------------------------------------------------------------
// zero a float region
__global__ void k_zero_f32(float* __restrict__ p, int n) {
    int i = blockIdx.x * blockDim.x + threadIdx.x;
    if (i < n) p[i] = 0.0f;
}

// convert the two big layer-1 weights [256 x 512] f32 -> bf16
__global__ void k_cvt_w1(const float* __restrict__ Wl1, const float* __restrict__ Wr1,
                         bf16_t* __restrict__ Wl1b, bf16_t* __restrict__ Wr1b) {
    int i = blockIdx.x * blockDim.x + threadIdx.x;
    const int sz = F_HID * F_IN;
    if (i < sz)            Wl1b[i]      = (bf16_t)Wl1[i];
    else if (i < 2 * sz)   Wr1b[i - sz] = (bf16_t)Wr1[i - sz];
}

// convert + zero-pad layer-2 weights [20 x 256] -> [32 x 256] bf16, and bias -> 32 f32
__global__ void k_cvt_w2(const float* __restrict__ Wl2, const float* __restrict__ Wr2,
                         const float* __restrict__ bl2,
                         bf16_t* __restrict__ Wl2b, bf16_t* __restrict__ Wr2b,
                         float* __restrict__ bl2p) {
    int i = blockIdx.x * blockDim.x + threadIdx.x;
    const int sz = F_OUT_PAD * F_HID; // 8192
    if (i < sz) {
        int r = i >> 8;
        Wl2b[i] = (r < F_OUT) ? (bf16_t)Wl2[i] : (bf16_t)0.0f;
    } else if (i < 2 * sz) {
        int j = i - sz; int r = j >> 8;
        Wr2b[j] = (r < F_OUT) ? (bf16_t)Wr2[j] : (bf16_t)0.0f;
    } else if (i < 2 * sz + F_OUT_PAD) {
        int j = i - 2 * sz;
        bl2p[j] = (j < F_OUT) ? bl2[j] : 0.0f;
    }
}

// convert x[:T1] f32 -> bf16
__global__ void k_cvt_x(const float* __restrict__ x, bf16_t* __restrict__ xb, int n) {
    int i = blockIdx.x * blockDim.x + threadIdx.x;
    if (i < n) xb[i] = (bf16_t)x[i];
}

// ------------------------------------------------------------------
// layer-1 edge scatter: 128 threads per edge, 4 floats each
__global__ void k_scatter1(const float* __restrict__ x,
                           const int* __restrict__ src, const int* __restrict__ dst,
                           float* __restrict__ agg, float* __restrict__ cnt) {
    int tid = blockIdx.x * blockDim.x + threadIdx.x;  // E1*128 threads
    int e = tid >> 7;
    if (e >= E1) return;
    int f = (tid & 127) << 2;
    int s = src[e], d = dst[e];
    const float4 v = *(const float4*)(x + (size_t)s * F_IN + f);
    float* p = agg + (size_t)d * F_IN + f;
    atomicAdd(p + 0, v.x);
    atomicAdd(p + 1, v.y);
    atomicAdd(p + 2, v.z);
    atomicAdd(p + 3, v.w);
    if (f == 0) atomicAdd(cnt + d, 1.0f);
}

// layer-2 edge scatter: 64 threads per edge, 4 bf16->f32 each
struct bf4 { bf16_t a, b, c, d; };
__global__ void k_scatter2(const bf16_t* __restrict__ h,
                           const int* __restrict__ src, const int* __restrict__ dst,
                           float* __restrict__ agg, float* __restrict__ cnt) {
    int tid = blockIdx.x * blockDim.x + threadIdx.x;  // E2*64 threads
    int e = tid >> 6;
    if (e >= E2) return;
    int f = (tid & 63) << 2;
    int s = src[e], d = dst[e];
    bf4 v = *(const bf4*)(h + (size_t)s * F_HID + f);
    float* p = agg + (size_t)d * F_HID + f;
    atomicAdd(p + 0, (float)v.a);
    atomicAdd(p + 1, (float)v.b);
    atomicAdd(p + 2, (float)v.c);
    atomicAdd(p + 3, (float)v.d);
    if (f == 0) atomicAdd(cnt + d, 1.0f);
}

// mean = agg / max(cnt,1), store bf16   (shift = log2(row width))
template<int SHIFT>
__global__ void k_mean(const float* __restrict__ agg, const float* __restrict__ cnt,
                       bf16_t* __restrict__ out, int n) {
    int i = blockIdx.x * blockDim.x + threadIdx.x;
    if (i >= n) return;
    int row = i >> SHIFT;
    float c = cnt[row];
    out[i] = (bf16_t)(agg[i] / fmaxf(c, 1.0f));
}

// ------------------------------------------------------------------
// WMMA K-loop helper. Atile: 16 rows (stride lda), Wtile: 16 rows (stride ldw),
// ksteps iterations of K=32.  Fragment layouts per CDNA5 ISA.
__device__ __forceinline__ void wmma_kloop(v8f& c,
                                           const bf16_t* __restrict__ Atile, int lda,
                                           const bf16_t* __restrict__ Wtile, int ldw,
                                           int ksteps, int lane) {
    const int m = lane & 15;
    const int g = lane >> 4;
    const bf16_t* arow = Atile + (size_t)m * lda + g * 8;   // K chunks {8g..8g+7},{8g+16..}
    const bf16_t* wrow = Wtile + (size_t)m * ldw + g * 8;   // col n == row n of W; K chunks {8g..8g+7},{8g+16..}
    for (int kt = 0; kt < ksteps; ++kt) {
        union { v16bf v; v8bf h[2]; } a, b;
        a.h[0] = *(const v8bf*)(arow + kt * 32);
        a.h[1] = *(const v8bf*)(arow + kt * 32 + 16);
        b.h[0] = *(const v8bf*)(wrow + kt * 32);
        b.h[1] = *(const v8bf*)(wrow + kt * 32 + 16);
        c = __builtin_amdgcn_wmma_f32_16x16x32_bf16(false, a.v, false, b.v,
                                                    (short)0, c, false, false);
    }
}

// layer-1 GEMM: h = relu(mean1 @ Wl1^T + bl1 + x_tgt @ Wr1^T), bf16 out [T1 x 256]
__global__ void k_gemm1(const bf16_t* __restrict__ mean1b, const bf16_t* __restrict__ xb,
                        const bf16_t* __restrict__ Wl1b, const bf16_t* __restrict__ Wr1b,
                        const float* __restrict__ bl1, bf16_t* __restrict__ h) {
    const int tilesN = F_HID / 16;                       // 16
    int wave = blockIdx.x * (blockDim.x >> 5) + (threadIdx.x >> 5);
    int mt = wave / tilesN;
    int nt = wave - mt * tilesN;
    if (mt >= T1 / 16) return;                            // wave-uniform
    int lane = threadIdx.x & 31;

    v8f c = {};
    wmma_kloop(c, mean1b + (size_t)mt * 16 * F_IN, F_IN,
               Wl1b + (size_t)nt * 16 * F_IN, F_IN, F_IN / 32, lane);
    wmma_kloop(c, xb + (size_t)mt * 16 * F_IN, F_IN,
               Wr1b + (size_t)nt * 16 * F_IN, F_IN, F_IN / 32, lane);

    int n = nt * 16 + (lane & 15);
    float bias = bl1[n];
    int mg = mt * 16 + ((lane >> 4) << 3);
#pragma unroll
    for (int r = 0; r < 8; ++r) {
        float v = c[r] + bias;
        v = v > 0.0f ? v : 0.0f;
        h[(size_t)(mg + r) * F_HID + n] = (bf16_t)v;
    }
}

// layer-2 GEMM: out = mean2 @ Wl2^T + bl2 + h[:T2] @ Wr2^T, f32 out [T2 x 20]
__global__ void k_gemm2(const bf16_t* __restrict__ mean2b, const bf16_t* __restrict__ hb,
                        const bf16_t* __restrict__ Wl2b, const bf16_t* __restrict__ Wr2b,
                        const float* __restrict__ bl2p, float* __restrict__ out) {
    const int tilesN = F_OUT_PAD / 16;                   // 2
    int wave = blockIdx.x * (blockDim.x >> 5) + (threadIdx.x >> 5);
    int mt = wave / tilesN;
    int nt = wave - mt * tilesN;
    if (mt >= T2 / 16) return;                            // wave-uniform
    int lane = threadIdx.x & 31;

    v8f c = {};
    wmma_kloop(c, mean2b + (size_t)mt * 16 * F_HID, F_HID,
               Wl2b + (size_t)nt * 16 * F_HID, F_HID, F_HID / 32, lane);
    wmma_kloop(c, hb + (size_t)mt * 16 * F_HID, F_HID,
               Wr2b + (size_t)nt * 16 * F_HID, F_HID, F_HID / 32, lane);

    int n = nt * 16 + (lane & 15);
    int mg = mt * 16 + ((lane >> 4) << 3);
    if (n < F_OUT) {                                      // divergence only after WMMAs
        float bias = bl2p[n];
#pragma unroll
        for (int r = 0; r < 8; ++r)
            out[(size_t)(mg + r) * F_OUT + n] = c[r] + bias;
    }
}

// per-row log-softmax over 20 classes, in place
__global__ void k_logsoftmax(float* __restrict__ out) {
    int row = blockIdx.x * blockDim.x + threadIdx.x;
    if (row >= T2) return;
    float* p = out + (size_t)row * F_OUT;
    float v[F_OUT];
    float m = -INFINITY;
#pragma unroll
    for (int i = 0; i < F_OUT; ++i) { v[i] = p[i]; m = fmaxf(m, v[i]); }
    float s = 0.0f;
#pragma unroll
    for (int i = 0; i < F_OUT; ++i) s += expf(v[i] - m);
    float lse = m + logf(s);
#pragma unroll
    for (int i = 0; i < F_OUT; ++i) p[i] = v[i] - lse;
}

// ------------------------------------------------------------------
static inline size_t align_up(size_t x, size_t a) { return (x + a - 1) & ~(a - 1); }

extern "C" void kernel_launch(void* const* d_in, const int* in_sizes, int n_in,
                              void* d_out, int out_size, void* d_ws, size_t ws_size,
                              hipStream_t stream) {
    (void)in_sizes; (void)n_in; (void)out_size; (void)ws_size;
    const float* x    = (const float*)d_in[0];
    const int*   src1 = (const int*)  d_in[1];
    const int*   dst1 = (const int*)  d_in[2];
    const int*   src2 = (const int*)  d_in[3];
    const int*   dst2 = (const int*)  d_in[4];
    const float* Wl1  = (const float*)d_in[5];
    const float* bl1  = (const float*)d_in[6];
    const float* Wr1  = (const float*)d_in[7];
    const float* Wl2  = (const float*)d_in[8];
    const float* bl2  = (const float*)d_in[9];
    const float* Wr2  = (const float*)d_in[10];
    float* out = (float*)d_out;

    // ---- workspace carve-up (f32 accumulators first, contiguous, zeroed each call)
    char* ws = (char*)d_ws;
    size_t off = 0;
    float* agg1 = (float*)(ws + off); off += (size_t)T1 * F_IN * 4;
    float* cnt1 = (float*)(ws + off); off += (size_t)T1 * 4;
    float* agg2 = (float*)(ws + off); off += (size_t)T2 * F_HID * 4;
    float* cnt2 = (float*)(ws + off); off += (size_t)T2 * 4;
    const int zeroLen = T1 * F_IN + T1 + T2 * F_HID + T2;

    off = align_up(off, 256);
    bf16_t* xb     = (bf16_t*)(ws + off); off += (size_t)T1 * F_IN * 2;
    bf16_t* mean1b = (bf16_t*)(ws + off); off += (size_t)T1 * F_IN * 2;
    bf16_t* hb     = (bf16_t*)(ws + off); off += (size_t)T1 * F_HID * 2;
    bf16_t* mean2b = (bf16_t*)(ws + off); off += (size_t)T2 * F_HID * 2;
    bf16_t* Wl1b   = (bf16_t*)(ws + off); off += (size_t)F_HID * F_IN * 2;
    bf16_t* Wr1b   = (bf16_t*)(ws + off); off += (size_t)F_HID * F_IN * 2;
    bf16_t* Wl2b   = (bf16_t*)(ws + off); off += (size_t)F_OUT_PAD * F_HID * 2;
    bf16_t* Wr2b   = (bf16_t*)(ws + off); off += (size_t)F_OUT_PAD * F_HID * 2;
    off = align_up(off, 256);
    float*  bl2p   = (float*)(ws + off);  off += F_OUT_PAD * 4;

    const int B = 256;

    // 1) zero accumulators
    k_zero_f32<<<(zeroLen + B - 1) / B, B, 0, stream>>>(agg1, zeroLen);
    // 2) weight / input conversions
    k_cvt_w1<<<(2 * F_HID * F_IN + B - 1) / B, B, 0, stream>>>(Wl1, Wr1, Wl1b, Wr1b);
    k_cvt_w2<<<(2 * F_OUT_PAD * F_HID + F_OUT_PAD + B - 1) / B, B, 0, stream>>>(
        Wl2, Wr2, bl2, Wl2b, Wr2b, bl2p);
    k_cvt_x<<<((T1 * F_IN) + B - 1) / B, B, 0, stream>>>(x, xb, T1 * F_IN);
    // 3) layer-1 aggregation (mean via f32 atomics in L2)
    k_scatter1<<<(E1 * 128) / B, B, 0, stream>>>(x, src1, dst1, agg1, cnt1);
    k_mean<9><<<((T1 * F_IN) + B - 1) / B, B, 0, stream>>>(agg1, cnt1, mean1b, T1 * F_IN);
    // 4) layer-1 WMMA GEMM + bias + relu -> h (bf16)
    {
        int tiles = (T1 / 16) * (F_HID / 16);             // 30000
        int wavesPerBlock = B / 32;                       // 8
        k_gemm1<<<tiles / wavesPerBlock, B, 0, stream>>>(mean1b, xb, Wl1b, Wr1b, bl1, hb);
    }
    // 5) layer-2 aggregation
    k_scatter2<<<(E2 * 64) / B, B, 0, stream>>>(hb, src2, dst2, agg2, cnt2);
    k_mean<8><<<((T2 * F_HID) + B - 1) / B, B, 0, stream>>>(agg2, cnt2, mean2b, T2 * F_HID);
    // 6) layer-2 WMMA GEMM + bias -> out (f32, masked to 20 cols)
    {
        int tiles = (T2 / 16) * (F_OUT_PAD / 16);         // 1024
        int wavesPerBlock = B / 32;
        k_gemm2<<<tiles / wavesPerBlock, B, 0, stream>>>(mean2b, hb, Wl2b, Wr2b, bl2p, out);
    }
    // 7) log-softmax in place
    k_logsoftmax<<<(T2 + B - 1) / B, B, 0, stream>>>(out);
}
